// FlashMHA_77043123356253
// MI455X (gfx1250) — matrix operand order repaired and checked by reading, not compiled, not run on hardware
//
#include <hip/hip_runtime.h>
#include <hip/hip_bf16.h>

// ---- problem constants --------------------------------------------------
#define B_  4
#define L_  2048
#define D_  1024
#define H_  16
#define DH_ 64
#define M_  (B_ * L_)        // 8192 rows of x
#define N_QKV (3 * D_)       // 3072

typedef __attribute__((ext_vector_type(16))) __bf16 v16bf;
typedef __attribute__((ext_vector_type(8)))  __bf16 v8bf;
typedef __attribute__((ext_vector_type(8)))  float  v8f;

// ---- WMMA helpers -------------------------------------------------------
static __device__ inline v8f wmma_bf16(v16bf a, v16bf b, v8f c) {
  // D = A(16x32 bf16) * B(32x16 bf16) + C(16x16 f32)
  return __builtin_amdgcn_wmma_f32_16x16x32_bf16(
      /*neg_a=*/false, a, /*neg_b=*/false, b,
      /*c_mod=*/(short)0, c, /*reuse_a=*/false, /*reuse_b=*/false);
}

// A fragment (16x32 bf16). rowPtr = &A[row_of_this_lane][k0] with k contiguous.
// Lane layout: row = lane&15; halves 0..15 -> k = 16g..16g+15 contiguous
// where g = lane>>4 (matches load_b_frag).
static __device__ inline v16bf load_a_frag(const __bf16* rowPtr, int g) {
  union { v16bf v; struct { v8bf lo, hi; } s; } u;
  u.s.lo = *(const v8bf*)(rowPtr + 16 * g);
  u.s.hi = *(const v8bf*)(rowPtr + 16 * g + 8);
  return u.v;
}

// B fragment (32x16 bf16) from a matrix stored as rows of N with k contiguous
// (i.e. W[n][k]): lane holds column n = lane&15, k = 16g .. 16g+15 contiguous.
static __device__ inline v16bf load_b_frag(const __bf16* p) {
  return *(const v16bf*)p;
}

// ---- fp32 -> bf16 cast --------------------------------------------------
__global__ __launch_bounds__(256) void cast_f32_bf16(
    const float* __restrict__ src, __bf16* __restrict__ dst, int n) {
  int i = (blockIdx.x * blockDim.x + threadIdx.x) * 4;
  if (i < n) {
    float4 f = *(const float4*)(src + i);
    dst[i + 0] = (__bf16)f.x;
    dst[i + 1] = (__bf16)f.y;
    dst[i + 2] = (__bf16)f.z;
    dst[i + 3] = (__bf16)f.w;
  }
}

// ---- QKV projection: [8192,1024] x [3072,1024]^T, scatter to Q/K/V ------
// Q is pre-scaled by 1/sqrt(DH) = 0.125 (exact power of two) so the
// attention kernel does no per-iteration score scaling.
__global__ __launch_bounds__(256) void gemm_qkv_kernel(
    const __bf16* __restrict__ X, const __bf16* __restrict__ W,
    __bf16* __restrict__ Qb, __bf16* __restrict__ Kb, __bf16* __restrict__ Vb) {
  const int K    = D_;
  const int lane = threadIdx.x & 31;
  const int g    = lane >> 4, r = lane & 15;
  const int wid  = threadIdx.x >> 5;
  const int wm   = wid >> 2, wn = wid & 3;          // 2x4 waves -> 128x256 block tile
  const int bm   = blockIdx.x & 63;                 // 64 m-blocks of 128
  const int bn   = blockIdx.x >> 6;                 // 12 n-blocks of 256
  const int m0   = bm * 128 + wm * 64;
  const int n0   = bn * 256 + wn * 64;

  v8f acc[4][4] = {};
  for (int k0 = 0; k0 < K; k0 += 32) {
    if (k0 + 32 < K) {  // next-step prefetch (global_prefetch_b8)
      __builtin_prefetch(X + (size_t)(m0 + r) * K + k0 + 32, 0, 1);
      __builtin_prefetch(W + (size_t)(n0 + r) * K + k0 + 32, 0, 1);
    }
    v16bf a[4];
#pragma unroll
    for (int i = 0; i < 4; i++)
      a[i] = load_a_frag(X + (size_t)(m0 + 16 * i + r) * K + k0, g);
#pragma unroll
    for (int j = 0; j < 4; j++) {
      v16bf b = load_b_frag(W + (size_t)(n0 + 16 * j + r) * K + k0 + 16 * g);
#pragma unroll
      for (int i = 0; i < 4; i++) acc[i][j] = wmma_bf16(a[i], b, acc[i][j]);
    }
  }

  // scatter: n -> (which, h, dh), m -> (b, l); dest layout [B,H,L,DH]
#pragma unroll
  for (int j = 0; j < 4; j++) {
    const int n     = n0 + 16 * j + r;
    const int which = n >> 10;
    const int h     = (n & 1023) >> 6;
    const int dh    = n & 63;
    const float sc  = (which == 0) ? 0.125f : 1.0f;   // fold attention scale into Q
    __bf16* dst = (which == 0) ? Qb : (which == 1) ? Kb : Vb;
#pragma unroll
    for (int i = 0; i < 4; i++) {
#pragma unroll
      for (int t = 0; t < 8; t++) {
        const int m  = m0 + 16 * i + t + 8 * g;   // C frag row = t + 8g
        const int bb = m >> 11;                   // / L_
        const int l  = m & 2047;
        dst[(((size_t)(bb * H_ + h)) * L_ + l) * DH_ + dh] = (__bf16)(acc[i][j][t] * sc);
      }
    }
  }
}

// ---- flash attention: per wave 16 query rows, online softmax ------------
// Block = 8 waves covering 128 query rows of one (b,h). All waves share the
// same K/V stream, so the 32x64 V tile is cooperatively transposed into LDS
// each step; P is transposed through per-wave LDS into A-fragment layout.
__global__ __launch_bounds__(256) void attn_kernel(
    const __bf16* __restrict__ Q, const __bf16* __restrict__ K,
    const __bf16* __restrict__ V, __bf16* __restrict__ ctx) {
  __shared__ __bf16 ldsP[8 * 16 * 32];            // per-wave 16x32 P tile (8 KB)
  __shared__ __bf16 ldsVt[64 * 32];               // Vt[dh][key] for this k-step (4 KB)

  const int tid  = threadIdx.x;
  const int lane = tid & 31;
  const int g    = lane >> 4, r = lane & 15;
  const int wid  = tid >> 5;
  const int qblocks = L_ / 128;                   // 16
  const int bh   = blockIdx.x / qblocks;          // b*H + h
  const int q0   = (blockIdx.x % qblocks) * 128 + wid * 16;
  const int b    = bh >> 4;
  const int h    = bh & 15;

  const __bf16* Qp = Q + (size_t)bh * L_ * DH_;
  const __bf16* Kp = K + (size_t)bh * L_ * DH_;
  const __bf16* Vp = V + (size_t)bh * L_ * DH_;
  __bf16* myP = ldsP + wid * 512;

  // cooperative V staging ids: 256 threads x 8 bf16 = 32x64 tile
  const int vkey   = tid >> 3;                    // 0..31 (key within tile)
  const int vchunk = tid & 7;                     // 0..7  (8-wide dh chunk)

  // Q fragments for the whole loop (DH=64 -> two 16x32 A frags); Q pre-scaled.
  v16bf qf[2];
  qf[0] = load_a_frag(Qp + (size_t)(q0 + r) * DH_ + 0,  g);
  qf[1] = load_a_frag(Qp + (size_t)(q0 + r) * DH_ + 32, g);

  float mrow[8], lrow[8];
#pragma unroll
  for (int i = 0; i < 8; i++) { mrow[i] = -3.0e38f; lrow[i] = 0.0f; }
  v8f o[4] = {};

  for (int k0 = 0; k0 < L_; k0 += 32) {
    // ---- cooperative V tile transpose into LDS: Vt[dh][key] ----
    {
      v8bf vrow = *(const v8bf*)(Vp + (size_t)(k0 + vkey) * DH_ + vchunk * 8);
#pragma unroll
      for (int e = 0; e < 8; e++)
        ldsVt[(vchunk * 8 + e) * 32 + vkey] = vrow[e];
    }

    // ---- S tiles: keys [k0,k0+16) and [k0+16,k0+32) ----
    v8f s0 = {}, s1 = {};
#pragma unroll
    for (int t = 0; t < 2; t++) {
      v16bf kb0 = load_b_frag(Kp + (size_t)(k0 + r) * DH_      + 32 * t + 16 * g);
      v16bf kb1 = load_b_frag(Kp + (size_t)(k0 + 16 + r) * DH_ + 32 * t + 16 * g);
      s0 = wmma_bf16(qf[t], kb0, s0);
      s1 = wmma_bf16(qf[t], kb1, s1);
    }

    // ---- online softmax: row stats per (vgpr i, half g); reduce over 16 lanes
    float alpha[8];
#pragma unroll
    for (int i = 0; i < 8; i++) {
      float v = fmaxf(s0[i], s1[i]);
#pragma unroll
      for (int mks = 1; mks < 16; mks <<= 1) v = fmaxf(v, __shfl_xor(v, mks, 32));
      float mn = fmaxf(mrow[i], v);
      alpha[i] = __expf(mrow[i] - mn);

      float p0 = __expf(s0[i] - mn);
      float p1 = __expf(s1[i] - mn);
      float rs = p0 + p1;
#pragma unroll
      for (int mks = 1; mks < 16; mks <<= 1) rs += __shfl_xor(rs, mks, 32);
      lrow[i] = lrow[i] * alpha[i] + rs;
      mrow[i] = mn;

      // P into LDS: row = i + 8g (C layout), cols r and 16+r
      myP[(i + 8 * g) * 32 + r]      = (__bf16)p0;
      myP[(i + 8 * g) * 32 + 16 + r] = (__bf16)p1;
    }

    // rescale O accumulators by alpha (per row = element index)
#pragma unroll
    for (int j = 0; j < 4; j++) {
#pragma unroll
      for (int i = 0; i < 8; i++) o[j][i] *= alpha[i];
    }

    __syncthreads();   // P + Vt visible

    // P as A fragment (cross-lane transpose via LDS)
    v16bf pa = load_a_frag(myP + r * 32, g);

    // O += P(16x32) * V(32x64): four 32x16 B frags straight from Vt
#pragma unroll
    for (int j = 0; j < 4; j++) {
      v16bf vb = load_b_frag(ldsVt + (16 * j + r) * 32 + 16 * g);
      o[j] = wmma_bf16(pa, vb, o[j]);
    }

    __syncthreads();   // protect LDS against next iteration's writes
  }

  // epilogue: normalize and write ctx as [B*L, D] bf16 (row = b*L+q, col = h*64+dh)
#pragma unroll
  for (int j = 0; j < 4; j++) {
#pragma unroll
    for (int i = 0; i < 8; i++) {
      float val = o[j][i] / lrow[i];
      size_t row = (size_t)b * L_ + q0 + i + 8 * g;
      ctx[row * D_ + h * DH_ + j * 16 + r] = (__bf16)val;
    }
  }
}

// ---- output projection: [8192,1024] x [1024,1024]^T -> f32 out ----------
__global__ __launch_bounds__(256) void gemm_proj_kernel(
    const __bf16* __restrict__ Xc, const __bf16* __restrict__ W,
    float* __restrict__ out) {
  const int K    = D_;
  const int lane = threadIdx.x & 31;
  const int g    = lane >> 4, r = lane & 15;
  const int wid  = threadIdx.x >> 5;
  const int wm   = wid >> 2, wn = wid & 3;
  const int bm   = blockIdx.x & 63;               // 64 m-blocks of 128
  const int bn   = blockIdx.x >> 6;               // 4 n-blocks of 256
  const int m0   = bm * 128 + wm * 64;
  const int n0   = bn * 256 + wn * 64;

  v8f acc[4][4] = {};
  for (int k0 = 0; k0 < K; k0 += 32) {
    if (k0 + 32 < K) {
      __builtin_prefetch(Xc + (size_t)(m0 + r) * K + k0 + 32, 0, 1);
      __builtin_prefetch(W + (size_t)(n0 + r) * K + k0 + 32, 0, 1);
    }
    v16bf a[4];
#pragma unroll
    for (int i = 0; i < 4; i++)
      a[i] = load_a_frag(Xc + (size_t)(m0 + 16 * i + r) * K + k0, g);
#pragma unroll
    for (int j = 0; j < 4; j++) {
      v16bf bfr = load_b_frag(W + (size_t)(n0 + 16 * j + r) * K + k0 + 16 * g);
#pragma unroll
      for (int i = 0; i < 4; i++) acc[i][j] = wmma_bf16(a[i], bfr, acc[i][j]);
    }
  }
#pragma unroll
  for (int j = 0; j < 4; j++) {
    const int n = n0 + 16 * j + r;
#pragma unroll
    for (int i = 0; i < 4; i++) {
#pragma unroll
      for (int t = 0; t < 8; t++) {
        const int m = m0 + 16 * i + t + 8 * g;
        out[(size_t)m * D_ + n] = acc[i][j][t];
      }
    }
  }
}

// ---- host-side launch ---------------------------------------------------
extern "C" void kernel_launch(void* const* d_in, const int* in_sizes, int n_in,
                              void* d_out, int out_size, void* d_ws, size_t ws_size,
                              hipStream_t stream) {
  (void)in_sizes; (void)n_in; (void)out_size; (void)ws_size;
  const float* x      = (const float*)d_in[0];
  const float* w_qkv  = (const float*)d_in[1];
  const float* w_proj = (const float*)d_in[2];
  float* out = (float*)d_out;

  char* ws = (char*)d_ws;
  size_t off = 0;
  __bf16* xb     = (__bf16*)(ws + off); off += (size_t)M_ * D_ * 2;
  __bf16* wqkvb  = (__bf16*)(ws + off); off += (size_t)N_QKV * D_ * 2;
  __bf16* wprojb = (__bf16*)(ws + off); off += (size_t)D_ * D_ * 2;
  __bf16* Qb     = (__bf16*)(ws + off); off += (size_t)B_ * H_ * L_ * DH_ * 2;
  __bf16* Kb     = (__bf16*)(ws + off); off += (size_t)B_ * H_ * L_ * DH_ * 2;
  __bf16* Vb     = (__bf16*)(ws + off); off += (size_t)B_ * H_ * L_ * DH_ * 2;
  __bf16* ctxb   = (__bf16*)(ws + off); off += (size_t)M_ * D_ * 2;

  const int nx = M_ * D_;          // 8388608
  const int nq = N_QKV * D_;       // 3145728
  const int np = D_ * D_;          // 1048576
  cast_f32_bf16<<<(nx / 4 + 255) / 256, 256, 0, stream>>>(x, xb, nx);
  cast_f32_bf16<<<(nq / 4 + 255) / 256, 256, 0, stream>>>(w_qkv, wqkvb, nq);
  cast_f32_bf16<<<(np / 4 + 255) / 256, 256, 0, stream>>>(w_proj, wprojb, np);

  gemm_qkv_kernel<<<64 * 12, 256, 0, stream>>>(xb, wqkvb, Qb, Kb, Vb);
  attn_kernel<<<B_ * H_ * (L_ / 128), 256, 0, stream>>>(Qb, Kb, Vb, ctxb);
  gemm_proj_kernel<<<64 * 4, 256, 0, stream>>>(ctxb, wprojb, out);
}
